// MDLRNN_83270825935633
// MI455X (gfx1250) — matrix operand order repaired and checked by reading, not compiled, not run on hardware
//
#include <hip/hip_runtime.h>

// MDLRNN persistent recurrence kernel for gfx1250 (MI455X).
// B=64, T=2048, I=128, H=256, M=256, O=128.
// 4 workgroups (16-batch-row tiles) x 8 waves; bf16 WMMA 16x16x32, f32 accum.
// Latency-oriented: split WMMA accumulation chains, native tanh, double-buffered
// X staging prefetched under the phase-2/3 WMMAs, 2 barriers per step.

#define B_  64
#define T_  2048
#define I_  128
#define H_  256
#define M_  256
#define O_  128

typedef __attribute__((ext_vector_type(16))) __bf16 v16bf;
typedef __attribute__((ext_vector_type(8)))  float  v8f;

__device__ __forceinline__ unsigned short f32_to_bf16_rne(float f) {
  unsigned u = __float_as_uint(f);
  unsigned r = u + 0x7FFFu + ((u >> 16) & 1u);
  return (unsigned short)(r >> 16);
}

__device__ __forceinline__ unsigned pack_bf16x2(float a, float b) {
  return (unsigned)f32_to_bf16_rne(a) | ((unsigned)f32_to_bf16_rne(b) << 16);
}

__device__ __forceinline__ float fast_tanh(float x) {
#if __has_builtin(__builtin_amdgcn_tanhf)
  return __builtin_amdgcn_tanhf(x);          // v_tanh_f32 (gfx1250 trans op)
#else
  // tanh(x) = 1 - 2/(1 + 2^(2*log2e*x)); clamp so exp2 can't overflow to NaN.
  float z  = fminf(fmaxf(x, -9.0f), 9.0f) * 2.885390081777927f;
  float e2 = __builtin_amdgcn_exp2f(z);
  return 1.0f - 2.0f * __builtin_amdgcn_rcpf(e2 + 1.0f);
#endif
}

// ---------------------------------------------------------------------------
// Prep: convert the four f32 weight matrices into bf16 B-fragment layout.
// B-fragment (16-bit, 32x16 per K-slice): lane<16 -> col N=lane, K={0..7,16..23};
// lane>=16 -> col N=lane-16, K={8..15,24..31}; each VGPR holds a consecutive K pair (even K in low half).
// Fragment storage: frag f, lane l, dword d -> ushort[((f*32+l)*8+d)*2 + b0].
// ws layout (ushort units):
//   [0)       W_ih : 16 Ntiles * 4 Kfrags * 512 = 32768
//   [32768)   W_mh : 16 * 8 * 512               = 65536
//   [98304)   W_ho :  8 * 8 * 512               = 32768
//   [131072)  W_hm : 16 * 8 * 512               = 65536   (total 196608 = 384KB)
// ---------------------------------------------------------------------------
__global__ __launch_bounds__(256) void prep_weights(
    const float* __restrict__ Wih, const float* __restrict__ Wmh,
    const float* __restrict__ Who, const float* __restrict__ Whm,
    unsigned short* __restrict__ ws) {
  int e = blockIdx.x * 256 + threadIdx.x;
  const int S0 = 32768, S1 = 65536, S2 = 32768, S3 = 65536;
  const float* W;
  int Kfrags, Ncols, base, le;
  if (e < S0)                { W = Wih; Kfrags = 4; Ncols = H_; base = 0;        le = e; }
  else if (e < S0+S1)        { W = Wmh; Kfrags = 8; Ncols = H_; base = S0;       le = e - S0; }
  else if (e < S0+S1+S2)     { W = Who; Kfrags = 8; Ncols = O_; base = S0+S1;    le = e - S0 - S1; }
  else if (e < S0+S1+S2+S3)  { W = Whm; Kfrags = 8; Ncols = M_; base = S0+S1+S2; le = e - S0 - S1 - S2; }
  else return;
  int b0 = le & 1;
  int d  = (le >> 1) & 7;
  int l  = (le >> 4) & 31;
  int f  = le >> 9;
  int nt = f / Kfrags, kf = f % Kfrags;
  int n = nt * 16 + (l & 15);
  int k = kf * 32 + ((d >> 2) << 4) + ((l >> 4) << 3) + ((d & 3) << 1) + b0;
  ws[base + le] = f32_to_bf16_rne(W[k * Ncols + n]);
}

// A-fragment index: element (row r in 0..15, K index k) of a 16xK bf16 A
// matrix -> ushort index within the A-frag array.
// 16-bit A 16x32: lanes 0-15 hold K {0..7,16..23}, lanes 16-31 hold {8..15,24..31}.
__device__ __forceinline__ int a_frag_idx(int r, int k) {
  int kf  = k >> 5;
  int k32 = k & 31;
  int b0  = k32 & 1;
  int dd  = ((k32 >> 4) << 2) | ((k32 >> 1) & 3);
  int hf  = (k32 >> 3) & 1;
  int dl  = (hf << 4) | r;
  return ((kf * 32 + dl) * 8 + dd) * 2 + b0;
}

// Store 4 consecutive K values (kb % 4 == 0) as two packed bf16 dwords.
// For kb aligned to 4, the two dwords are adjacent and 8B-aligned.
__device__ __forceinline__ void store_frag_quad(unsigned short* base, int r, int kb,
                                                float x, float y, float z, float w) {
  int d0 = a_frag_idx(r, kb) >> 1;  // even dword index
  uint2 v;
  v.x = pack_bf16x2(x, y);
  v.y = pack_bf16x2(z, w);
  *(uint2*)((unsigned*)base + d0) = v;   // ds_store_b64
}

__device__ __forceinline__ v8f wmma_bf16(v16bf a, v16bf b, v8f c) {
  return __builtin_amdgcn_wmma_f32_16x16x32_bf16(false, a, false, b,
                                                 (short)0, c, false, false);
}

// ---------------------------------------------------------------------------
// Main persistent kernel: one block per 16-row batch tile, 8 waves.
// Dynamic LDS (ushort units):
//   sWmh  [0,      65536)   W_mh fragments (128KB)
//   sWhm  [65536, 131072)   W_hm fragments (128KB)
//   sxA0  [131072,133120)   X_t   A-frags, buffer 0 (4KB)
//   sxA1  [133120,135168)   X_t   A-frags, buffer 1 (4KB)
//   smemA [135168,139264)   mem   A-frags (8KB)
//   shA   [139264,143360)   h     A-frags (8KB)
// Total 143360 ushorts = 280KB of the WGP's 320KB.
// ---------------------------------------------------------------------------
__global__ __launch_bounds__(256) void mdlrnn_main(
    const float* __restrict__ X, const float* __restrict__ mem0,
    const unsigned short* __restrict__ ws,
    const float* __restrict__ bh, const float* __restrict__ bmh,
    const float* __restrict__ bo, const float* __restrict__ bm,
    float* __restrict__ out) {
  extern __shared__ unsigned short lds[];
  unsigned short* sWmh  = lds;
  unsigned short* sWhm  = lds + 65536;
  unsigned short* sxA0  = lds + 131072;   // buffer stride 2048 ushorts
  unsigned short* smemA = lds + 135168;
  unsigned short* shA   = lds + 139264;

  const unsigned short* gWih = ws;
  const unsigned short* gWho = ws + 98304;

  const int tid  = threadIdx.x;
  const int lane = tid & 31;
  const int wv   = tid >> 5;          // wave 0..7
  const int bb   = blockIdx.x * 16;   // batch-tile base row
  const size_t OUT_MEM = (size_t)B_ * T_ * O_;

  // Copy hot weights into LDS (coalesced b128).
  for (int i = tid; i < 65536 / 8; i += 256) {
    ((uint4*)sWmh)[i] = ((const uint4*)(ws + 32768))[i];
    ((uint4*)sWhm)[i] = ((const uint4*)(ws + 131072))[i];
  }
  // Initial memory -> bf16 A-fragments (float4 loads, packed b64 LDS stores).
  #pragma unroll
  for (int i = 0; i < 4; ++i) {                 // 16*256/4 = 1024 float4s
    int e4 = tid + i * 256;
    int r  = e4 >> 6;                           // 64 float4 per row
    int kb = (e4 & 63) << 2;
    float4 mv = *(const float4*)(mem0 + (size_t)(bb + r) * M_ + kb);
    store_frag_quad(smemA, r, kb, mv.x, mv.y, mv.z, mv.w);
  }
  // Stage X[0] into buffer 0.
  #pragma unroll
  for (int i = 0; i < 2; ++i) {                 // 16*128/4 = 512 float4s
    int e4 = tid + i * 256;
    int r  = e4 >> 5;                           // 32 float4 per row
    int kb = (e4 & 31) << 2;
    float4 xv = *(const float4*)(X + ((size_t)(bb + r) * T_ + 0) * I_ + kb);
    store_frag_quad(sxA0, r, kb, xv.x, xv.y, xv.z, xv.w);
  }
  __syncthreads();

  // Per-lane biases, hoisted out of the T loop.
  const int col = lane & 15;
  const int hn0 = wv * 2, hn1 = wv * 2 + 1;
  const float biasH0 = bh[hn0 * 16 + col] + bmh[hn0 * 16 + col];
  const float biasH1 = bh[hn1 * 16 + col] + bmh[hn1 * 16 + col];
  float biasO[3];
  #pragma unroll
  for (int ti = 0; ti < 3; ++ti) {
    int tile = wv * 3 + ti;
    biasO[ti] = (tile < 8) ? bo[tile * 16 + col] : bm[(tile - 8) * 16 + col];
  }

  for (int t = 0; t < T_; ++t) {
    const unsigned short* xbuf = sxA0 + (t & 1) * 2048;

    // ---- phase 1: h tiles (2 per wave); 3 independent WMMA chains each ----
    #pragma unroll
    for (int ti = 0; ti < 2; ++ti) {
      const int n = wv * 2 + ti;
      const float bias = ti ? biasH1 : biasH0;
      v8f accX  = {bias, bias, bias, bias, bias, bias, bias, bias};
      v8f accM0 = {};
      v8f accM1 = {};
      #pragma unroll
      for (int f = 0; f < 4; ++f) {
        v16bf a = *(const v16bf*)(xbuf + (f * 32 + lane) * 16);
        v16bf b = *(const v16bf*)(gWih + ((n * 4 + f) * 32 + lane) * 16);
        accX = wmma_bf16(a, b, accX);
      }
      #pragma unroll
      for (int f = 0; f < 4; ++f) {
        v16bf a = *(const v16bf*)(smemA + (f * 32 + lane) * 16);
        v16bf b = *(const v16bf*)(sWmh + ((n * 8 + f) * 32 + lane) * 16);
        accM0 = wmma_bf16(a, b, accM0);
      }
      #pragma unroll
      for (int f = 4; f < 8; ++f) {
        v16bf a = *(const v16bf*)(smemA + (f * 32 + lane) * 16);
        v16bf b = *(const v16bf*)(sWmh + ((n * 8 + f) * 32 + lane) * 16);
        accM1 = wmma_bf16(a, b, accM1);
      }
      // tanh + scatter h into A-fragment layout (C: N=lane&15, M=v+8*(lane>>4))
      const int hc = n * 16 + col;
      #pragma unroll
      for (int v = 0; v < 8; ++v) {
        float hval = fast_tanh(accX[v] + accM0[v] + accM1[v]);
        int r = v + ((lane >> 4) << 3);
        shA[a_frag_idx(r, hc)] = f32_to_bf16_rne(hval);
      }
    }
    __syncthreads();

    // ---- prefetch X[t+1] into registers (overlaps with phase 2/3 WMMAs) ----
    const bool have_next = (t + 1 < T_);
    float4 xpre[2];
    if (have_next) {
      #pragma unroll
      for (int i = 0; i < 2; ++i) {
        int e4 = tid + i * 256;
        int r  = e4 >> 5;
        int kb = (e4 & 31) << 2;
        xpre[i] = *(const float4*)(X + ((size_t)(bb + r) * T_ + (t + 1)) * I_ + kb);
      }
    }

    // ---- phase 2/3: y (tiles 0..7) and mem' (tiles 8..23), 3 per wave ----
    const bool last = (t == T_ - 1);
    #pragma unroll
    for (int ti = 0; ti < 3; ++ti) {
      const int tile = wv * 3 + ti;
      const float bias = biasO[ti];
      v8f acc0 = {bias, bias, bias, bias, bias, bias, bias, bias};
      v8f acc1 = {};
      if (tile < 8) {                      // y = h @ W_ho + b_o
        const int n = tile;
        #pragma unroll
        for (int f = 0; f < 4; ++f) {
          v16bf a = *(const v16bf*)(shA + (f * 32 + lane) * 16);
          v16bf b = *(const v16bf*)(gWho + ((n * 8 + f) * 32 + lane) * 16);
          acc0 = wmma_bf16(a, b, acc0);
        }
        #pragma unroll
        for (int f = 4; f < 8; ++f) {
          v16bf a = *(const v16bf*)(shA + (f * 32 + lane) * 16);
          v16bf b = *(const v16bf*)(gWho + ((n * 8 + f) * 32 + lane) * 16);
          acc1 = wmma_bf16(a, b, acc1);
        }
        const int oc = n * 16 + col;
        #pragma unroll
        for (int v = 0; v < 8; ++v) {
          int r = v + ((lane >> 4) << 3);
          out[((size_t)(bb + r) * T_ + t) * O_ + oc] = acc0[v] + acc1[v];
        }
      } else {                             // mem' = h @ W_hm + b_m
        const int n = tile - 8;
        #pragma unroll
        for (int f = 0; f < 4; ++f) {
          v16bf a = *(const v16bf*)(shA + (f * 32 + lane) * 16);
          v16bf b = *(const v16bf*)(sWhm + ((n * 8 + f) * 32 + lane) * 16);
          acc0 = wmma_bf16(a, b, acc0);
        }
        #pragma unroll
        for (int f = 4; f < 8; ++f) {
          v16bf a = *(const v16bf*)(shA + (f * 32 + lane) * 16);
          v16bf b = *(const v16bf*)(sWhm + ((n * 8 + f) * 32 + lane) * 16);
          acc1 = wmma_bf16(a, b, acc1);
        }
        const int mc = n * 16 + col;
        #pragma unroll
        for (int v = 0; v < 8; ++v) {
          float mval = acc0[v] + acc1[v];
          int r = v + ((lane >> 4) << 3);
          smemA[a_frag_idx(r, mc)] = f32_to_bf16_rne(mval);
          if (last)
            out[OUT_MEM + (size_t)(bb + r) * M_ + mc] = mval;
        }
      }
    }

    // ---- commit prefetched X[t+1] to the alternate buffer ----
    if (have_next) {
      unsigned short* nbuf = sxA0 + ((t + 1) & 1) * 2048;
      #pragma unroll
      for (int i = 0; i < 2; ++i) {
        int e4 = tid + i * 256;
        int r  = e4 >> 5;
        int kb = (e4 & 31) << 2;
        store_frag_quad(nbuf, r, kb, xpre[i].x, xpre[i].y, xpre[i].z, xpre[i].w);
      }
    }
    __syncthreads();
  }
}

extern "C" void kernel_launch(void* const* d_in, const int* in_sizes, int n_in,
                              void* d_out, int out_size, void* d_ws, size_t ws_size,
                              hipStream_t stream) {
  (void)in_sizes; (void)n_in; (void)out_size; (void)ws_size;
  const float* X    = (const float*)d_in[0];
  const float* mem0 = (const float*)d_in[1];
  const float* Wih  = (const float*)d_in[2];
  const float* bh   = (const float*)d_in[3];
  const float* Wmh  = (const float*)d_in[4];
  const float* bmh  = (const float*)d_in[5];
  const float* Who  = (const float*)d_in[6];
  const float* bo   = (const float*)d_in[7];
  const float* Whm  = (const float*)d_in[8];
  const float* bm   = (const float*)d_in[9];
  float* out = (float*)d_out;
  unsigned short* ws = (unsigned short*)d_ws;

  // 196608 bf16 elements of fragment-swizzled weights.
  prep_weights<<<768, 256, 0, stream>>>(Wih, Wmh, Who, Whm, ws);

  const size_t shmem = (size_t)143360 * sizeof(unsigned short);  // 280 KB
  mdlrnn_main<<<4, 256, shmem, stream>>>(X, mem0, ws, bh, bmh, bo, bm, out);
}
